// TypeDecoder_73693048865000
// MI455X (gfx1250) — compile-verified, matrix operand order checked by reading, not run on hardware
//
#include <hip/hip_runtime.h>
#include <hip/hip_bf16.h>
#include <math.h>

typedef __bf16 bf16_t;
typedef __attribute__((ext_vector_type(16))) __bf16 v16bf;
typedef __attribute__((ext_vector_type(8)))  __bf16 v8bf;
typedef __attribute__((ext_vector_type(4)))  __bf16 v4bf;
typedef __attribute__((ext_vector_type(8)))  float  v8f;

#define BB 128
#define SS 512
#define HH 1024
#define VV 5000
#define LL 32
#define VP 5008          // V padded to a multiple of 16 for WMMA tiles
#define SOS 1

// ---------------------------------------------------------------------------
// WMMA fragment loader (16-bit A/B layout per CDNA5 ISA 7.12.2):
//   lanes 0-15  : row = lane,    K = k0+{0..7}  and k0+{16..23}
//   lanes 16-31 : row = lane-16, K = k0+{8..15} and k0+{24..31}
// Both A (rows of X) and B (rows of W, since C = X*W^T) use this loader.
// ---------------------------------------------------------------------------
__device__ __forceinline__ v16bf ldfrag(const bf16_t* __restrict__ base, int ld, int k0) {
  int lane = threadIdx.x & 31;
  const bf16_t* p = base + (size_t)(lane & 15) * ld + k0 + ((lane & 16) ? 8 : 0);
  v8bf lo = *reinterpret_cast<const v8bf*>(p);
  v8bf hi = *reinterpret_cast<const v8bf*>(p + 16);
  v16bf f;
#pragma unroll
  for (int i = 0; i < 8; ++i) { f[i] = lo[i]; f[i + 8] = hi[i]; }
  return f;
}

// Skinny GEMM: C[M,N] = A[M,K](bf16) * W[N,K](bf16)^T + bias[N], fp32 out.
// One wave per (16*MH) x 16 tile. K-loop unrolled x2 with ping-pong fragment
// buffers: loads for one buffer overlap WMMAs on the other, with no
// register-copy rotation (buffers alternate roles structurally).
// Requires K % 64 == 0 (true here: K in {1024, 2048}).
template <int MH>
__global__ __launch_bounds__(256) void k_gemm_xwt(
    const bf16_t* __restrict__ A, int lda,
    const bf16_t* __restrict__ W, int ldw,
    const float* __restrict__ bias, int biasN,
    float* __restrict__ C, int ldc,
    int K, int tilesN, int totalTiles)
{
  int wid = (int)((blockIdx.x * blockDim.x + threadIdx.x) >> 5);
  if (wid >= totalTiles) return;
  int mg = wid / tilesN, nt = wid % tilesN;
  const bf16_t* Ar = A + (size_t)mg * MH * 16 * lda;
  const bf16_t* Wr = W + (size_t)nt * 16 * ldw;

  v8f acc[MH];
#pragma unroll
  for (int mi = 0; mi < MH; ++mi) acc[mi] = (v8f){};

  // buf0 <- slice 0
  v16bf bw0 = ldfrag(Wr, ldw, 0);
  v16bf a0[MH];
#pragma unroll
  for (int mi = 0; mi < MH; ++mi) a0[mi] = ldfrag(Ar + (size_t)mi * 16 * lda, lda, 0);

  int k0 = 0;
  for (; k0 < K - 64; k0 += 64) {
    // buf1 <- k0+32 while buf0 WMMAs run
    v16bf bw1 = ldfrag(Wr, ldw, k0 + 32);
    v16bf a1[MH];
#pragma unroll
    for (int mi = 0; mi < MH; ++mi) a1[mi] = ldfrag(Ar + (size_t)mi * 16 * lda, lda, k0 + 32);
#pragma unroll
    for (int mi = 0; mi < MH; ++mi)
      acc[mi] = __builtin_amdgcn_wmma_f32_16x16x32_bf16(false, a0[mi], false, bw0,
                                                        (short)0, acc[mi], false, false);
    // buf0 <- k0+64 while buf1 WMMAs run
    bw0 = ldfrag(Wr, ldw, k0 + 64);
#pragma unroll
    for (int mi = 0; mi < MH; ++mi) a0[mi] = ldfrag(Ar + (size_t)mi * 16 * lda, lda, k0 + 64);
#pragma unroll
    for (int mi = 0; mi < MH; ++mi)
      acc[mi] = __builtin_amdgcn_wmma_f32_16x16x32_bf16(false, a1[mi], false, bw1,
                                                        (short)0, acc[mi], false, false);
  }
  // epilogue: slices k0 (in buf0) and k0+32
  {
    v16bf bw1 = ldfrag(Wr, ldw, k0 + 32);
    v16bf a1[MH];
#pragma unroll
    for (int mi = 0; mi < MH; ++mi) a1[mi] = ldfrag(Ar + (size_t)mi * 16 * lda, lda, k0 + 32);
#pragma unroll
    for (int mi = 0; mi < MH; ++mi)
      acc[mi] = __builtin_amdgcn_wmma_f32_16x16x32_bf16(false, a0[mi], false, bw0,
                                                        (short)0, acc[mi], false, false);
#pragma unroll
    for (int mi = 0; mi < MH; ++mi)
      acc[mi] = __builtin_amdgcn_wmma_f32_16x16x32_bf16(false, a1[mi], false, bw1,
                                                        (short)0, acc[mi], false, false);
  }

  // C/D layout: lane<16 -> col=lane, rows 0-7 ; lane>=16 -> col=lane-16, rows 8-15
  int lane = threadIdx.x & 31;
  int col  = nt * 16 + (lane & 15);
  float bv = (col < biasN) ? bias[col] : 0.0f;
#pragma unroll
  for (int mi = 0; mi < MH; ++mi) {
    int row0 = (mg * MH + mi) * 16 + ((lane & 16) ? 8 : 0);
#pragma unroll
    for (int v = 0; v < 8; ++v)
      C[(size_t)(row0 + v) * ldc + col] = acc[mi][v] + bv;
  }
}

// ---------------------------------------------------------------------------
// Precision / init kernels
// ---------------------------------------------------------------------------
__global__ void k_cvt(const float* __restrict__ s, bf16_t* __restrict__ d, int n) {
  int i = blockIdx.x * blockDim.x + threadIdx.x;
  if (i < n) d[i] = (bf16_t)s[i];
}

__global__ void k_pad_outw(const float* __restrict__ s, bf16_t* __restrict__ d) {
  int i = blockIdx.x * blockDim.x + threadIdx.x;  // over VP*HH
  if (i >= VP * HH) return;
  d[i] = (i < VV * HH) ? (bf16_t)s[i] : (bf16_t)0.0f;
}

__global__ void k_init_out(float* __restrict__ probs, float* __restrict__ sampled) {
  int i = blockIdx.x * blockDim.x + threadIdx.x;  // over BB*VV
  if (i >= BB * VV) return;
  int b = i / VV, v = i % VV;
  probs[(size_t)b * LL * VV + v] = (v == SOS) ? 1.0f : 0.0f;  // t = 0 one-hot
  if (i < BB) sampled[(size_t)i * LL + 0] = (float)SOS;
}

__global__ void k_init_state(float* __restrict__ h, bf16_t* __restrict__ h_bf,
                             int* __restrict__ idx) {
  int i = blockIdx.x * blockDim.x + threadIdx.x;  // over BB*HH
  if (i >= BB * HH) return;
  h[i] = 0.0f; h_bf[i] = (bf16_t)0.0f;
  if (i < BB) idx[i] = SOS;
}

// ---------------------------------------------------------------------------
// Attention (fp32-encoder fallback): scores -> softmax(S) -> ctx, fused;
// also gathers embedding and writes x_bf = [ctx | emb] (bf16).
// One block (256 thr = 8 waves) per batch row.
// ---------------------------------------------------------------------------
__global__ __launch_bounds__(256) void k_attn_f32(
    const float* __restrict__ enc, const float* __restrict__ th,
    const bf16_t* __restrict__ emb_bf, const int* __restrict__ idx,
    bf16_t* __restrict__ x_bf)
{
  int b = blockIdx.x, tid = threadIdx.x;
  int lane = tid & 31, wave = tid >> 5;
  __shared__ float sw[SS];
  __shared__ float red[256];
  const float* encb = enc + (size_t)b * SS * HH;
  const float* thb  = th + (size_t)b * HH;

  for (int s = wave; s < SS; s += 8) {
    const float* row = encb + (size_t)s * HH;
    float acc = 0.0f;
    for (int j = lane; j < HH; j += 32) acc += row[j] * thb[j];
#pragma unroll
    for (int off = 16; off > 0; off >>= 1) acc += __shfl_xor(acc, off, 32);
    if (lane == 0) sw[s] = acc;
  }
  __syncthreads();

  float m = -1e30f;
  for (int s = tid; s < SS; s += 256) m = fmaxf(m, sw[s]);
  red[tid] = m; __syncthreads();
  for (int off = 128; off > 0; off >>= 1) {
    if (tid < off) red[tid] = fmaxf(red[tid], red[tid + off]);
    __syncthreads();
  }
  m = red[0]; __syncthreads();
  float sum = 0.0f;
  for (int s = tid; s < SS; s += 256) { float e = __expf(sw[s] - m); sw[s] = e; sum += e; }
  red[tid] = sum; __syncthreads();
  for (int off = 128; off > 0; off >>= 1) {
    if (tid < off) red[tid] += red[tid + off];
    __syncthreads();
  }
  float inv = 1.0f / red[0];
  __syncthreads();

  int h0 = tid * 4;
  float4 acc4 = make_float4(0.f, 0.f, 0.f, 0.f);
  for (int s = 0; s < SS; ++s) {
    float wv = sw[s];
    float4 e4 = *reinterpret_cast<const float4*>(encb + (size_t)s * HH + h0);
    acc4.x += wv * e4.x; acc4.y += wv * e4.y;
    acc4.z += wv * e4.z; acc4.w += wv * e4.w;
  }
  bf16_t* xb = x_bf + (size_t)b * (2 * HH);
  xb[h0 + 0] = (bf16_t)(acc4.x * inv);
  xb[h0 + 1] = (bf16_t)(acc4.y * inv);
  xb[h0 + 2] = (bf16_t)(acc4.z * inv);
  xb[h0 + 3] = (bf16_t)(acc4.w * inv);

  const bf16_t* er = emb_bf + (size_t)idx[b] * HH;
  for (int hh = tid; hh < HH; hh += 256) xb[HH + hh] = er[hh];
}

// ---------------------------------------------------------------------------
// Attention (bf16-encoder fast path): encoder copy is 134 MB -> L2-resident,
// halving the dominant memory traffic of the whole workload.
// ---------------------------------------------------------------------------
__global__ __launch_bounds__(256) void k_attn_bf16(
    const bf16_t* __restrict__ enc, const float* __restrict__ th,
    const bf16_t* __restrict__ emb_bf, const int* __restrict__ idx,
    bf16_t* __restrict__ x_bf)
{
  int b = blockIdx.x, tid = threadIdx.x;
  int lane = tid & 31, wave = tid >> 5;
  __shared__ float sw[SS];
  __shared__ float red[256];
  const bf16_t* encb = enc + (size_t)b * SS * HH;
  const float*  thb  = th + (size_t)b * HH;

  // scores[s] = enc[b,s,:] . th[b,:]   (4 bf16 = 8B per lane per iter)
  for (int s = wave; s < SS; s += 8) {
    const bf16_t* row = encb + (size_t)s * HH;
    float acc = 0.0f;
    for (int j = lane * 4; j < HH; j += 128) {
      v4bf e = *reinterpret_cast<const v4bf*>(row + j);
      acc += (float)e[0] * thb[j]     + (float)e[1] * thb[j + 1]
           + (float)e[2] * thb[j + 2] + (float)e[3] * thb[j + 3];
    }
#pragma unroll
    for (int off = 16; off > 0; off >>= 1) acc += __shfl_xor(acc, off, 32);
    if (lane == 0) sw[s] = acc;
  }
  __syncthreads();

  float m = -1e30f;
  for (int s = tid; s < SS; s += 256) m = fmaxf(m, sw[s]);
  red[tid] = m; __syncthreads();
  for (int off = 128; off > 0; off >>= 1) {
    if (tid < off) red[tid] = fmaxf(red[tid], red[tid + off]);
    __syncthreads();
  }
  m = red[0]; __syncthreads();
  float sum = 0.0f;
  for (int s = tid; s < SS; s += 256) { float e = __expf(sw[s] - m); sw[s] = e; sum += e; }
  red[tid] = sum; __syncthreads();
  for (int off = 128; off > 0; off >>= 1) {
    if (tid < off) red[tid] += red[tid + off];
    __syncthreads();
  }
  float inv = 1.0f / red[0];
  __syncthreads();

  // ctx[h] = sum_s w[s] * enc[b,s,h]
  int h0 = tid * 4;
  float4 acc4 = make_float4(0.f, 0.f, 0.f, 0.f);
  for (int s = 0; s < SS; ++s) {
    float wv = sw[s];
    v4bf e4 = *reinterpret_cast<const v4bf*>(encb + (size_t)s * HH + h0);
    acc4.x += wv * (float)e4[0]; acc4.y += wv * (float)e4[1];
    acc4.z += wv * (float)e4[2]; acc4.w += wv * (float)e4[3];
  }
  bf16_t* xb = x_bf + (size_t)b * (2 * HH);
  xb[h0 + 0] = (bf16_t)(acc4.x * inv);
  xb[h0 + 1] = (bf16_t)(acc4.y * inv);
  xb[h0 + 2] = (bf16_t)(acc4.z * inv);
  xb[h0 + 3] = (bf16_t)(acc4.w * inv);

  const bf16_t* er = emb_bf + (size_t)idx[b] * HH;
  for (int hh = tid; hh < HH; hh += 256) xb[HH + hh] = er[hh];
}

// ---------------------------------------------------------------------------
// GRU pointwise cell: h_new = (1-z)*n + z*h  (also refresh bf16 copy)
// ---------------------------------------------------------------------------
__global__ void k_gru_cell(const float* __restrict__ gi, const float* __restrict__ gh,
                           float* __restrict__ h, bf16_t* __restrict__ h_bf) {
  int i = blockIdx.x * blockDim.x + threadIdx.x;  // over BB*HH
  if (i >= BB * HH) return;
  int b = i / HH, n = i % HH;
  const float* gib = gi + (size_t)b * 3 * HH;
  const float* ghb = gh + (size_t)b * 3 * HH;
  float r  = 1.0f / (1.0f + __expf(-(gib[n] + ghb[n])));
  float z  = 1.0f / (1.0f + __expf(-(gib[HH + n] + ghb[HH + n])));
  float nn = tanhf(gib[2 * HH + n] + r * ghb[2 * HH + n]);
  float hv = h[i];
  float hn = (1.0f - z) * nn + z * hv;
  h[i] = hn; h_bf[i] = (bf16_t)hn;
}

// ---------------------------------------------------------------------------
// Vocab softmax + argmax (first-max tie-break). One block per batch row.
// exp values cached back into the scratch logits row to avoid recompute.
// ---------------------------------------------------------------------------
__global__ __launch_bounds__(256) void k_vsoftmax(
    float* __restrict__ logits, float* __restrict__ out_probs,
    float* __restrict__ out_idx, int* __restrict__ idx_ws, int t)
{
  int b = blockIdx.x, tid = threadIdx.x;
  float* lb = logits + (size_t)b * VP;
  __shared__ float red[256];
  __shared__ int  redi[256];
  float m = -1e30f; int mi = 0;
  for (int v = tid; v < VV; v += 256) { float x = lb[v]; if (x > m) { m = x; mi = v; } }
  red[tid] = m; redi[tid] = mi; __syncthreads();
  for (int off = 128; off > 0; off >>= 1) {
    if (tid < off) {
      float a = red[tid], c = red[tid + off];
      if (c > a || (c == a && redi[tid + off] < redi[tid])) { red[tid] = c; redi[tid] = redi[tid + off]; }
    }
    __syncthreads();
  }
  m = red[0]; int best = redi[0]; __syncthreads();
  float sum = 0.0f;
  for (int v = tid; v < VV; v += 256) { float e = __expf(lb[v] - m); lb[v] = e; sum += e; }
  red[tid] = sum; __syncthreads();
  for (int off = 128; off > 0; off >>= 1) {
    if (tid < off) red[tid] += red[tid + off];
    __syncthreads();
  }
  float inv = 1.0f / red[0];
  float* op = out_probs + ((size_t)b * LL + t) * VV;
  for (int v = tid; v < VV; v += 256) op[v] = lb[v] * inv;
  if (tid == 0) { idx_ws[b] = best; out_idx[(size_t)b * LL + t] = (float)best; }
}

// ---------------------------------------------------------------------------
// Host launcher
// ---------------------------------------------------------------------------
extern "C" void kernel_launch(void* const* d_in, const int* in_sizes, int n_in,
                              void* d_out, int out_size, void* d_ws, size_t ws_size,
                              hipStream_t stream) {
  const float* enc    = (const float*)d_in[0];
  const float* embed  = (const float*)d_in[1];
  const float* attn_w = (const float*)d_in[2];
  const float* attn_b = (const float*)d_in[3];
  const float* w_ih   = (const float*)d_in[4];
  const float* w_hh   = (const float*)d_in[5];
  const float* b_ih   = (const float*)d_in[6];
  const float* b_hh   = (const float*)d_in[7];
  const float* out_w  = (const float*)d_in[8];
  const float* out_b  = (const float*)d_in[9];

  // Workspace carve-out (~47 MB base; +134 MB optional bf16 encoder copy)
  char* p = (char*)d_ws;
  auto take = [&p](size_t bytes) { char* r = p; p += (bytes + 255) & ~(size_t)255; return r; };
  bf16_t* attnw_bf = (bf16_t*)take((size_t)HH * HH * 2);
  bf16_t* wih_bf   = (bf16_t*)take((size_t)3 * HH * 2 * HH * 2);
  bf16_t* whh_bf   = (bf16_t*)take((size_t)3 * HH * HH * 2);
  bf16_t* outw_bf  = (bf16_t*)take((size_t)VP * HH * 2);
  bf16_t* emb_bf   = (bf16_t*)take((size_t)VV * HH * 2);
  float*  h        = (float*) take((size_t)BB * HH * 4);
  bf16_t* h_bf     = (bf16_t*)take((size_t)BB * HH * 2);
  float*  th       = (float*) take((size_t)BB * HH * 4);
  bf16_t* x_bf     = (bf16_t*)take((size_t)BB * 2 * HH * 2);
  float*  gi       = (float*) take((size_t)BB * 3 * HH * 4);
  float*  gh       = (float*) take((size_t)BB * 3 * HH * 4);
  float*  logits   = (float*) take((size_t)BB * VP * 4);
  int*    idxb     = (int*)   take((size_t)BB * 4);
  size_t base_bytes = (size_t)(p - (char*)d_ws);
  const size_t enc_elems = (size_t)BB * SS * HH;
  // Deterministic per-run choice: ws_size is fixed across calls.
  bool use_enc_bf = (ws_size >= base_bytes + enc_elems * 2 + 4096);
  bf16_t* enc_bf = use_enc_bf ? (bf16_t*)take(enc_elems * 2) : nullptr;

  float* out_probs = (float*)d_out;
  float* out_idx   = out_probs + (size_t)BB * LL * VV;

  auto cvt = [&](const float* s, bf16_t* d, int n) {
    k_cvt<<<(unsigned)((n + 255) / 256), 256, 0, stream>>>(s, d, n);
  };
  cvt(attn_w, attnw_bf, HH * HH);
  cvt(w_ih,   wih_bf,   3 * HH * 2 * HH);
  cvt(w_hh,   whh_bf,   3 * HH * HH);
  cvt(embed,  emb_bf,   VV * HH);
  if (use_enc_bf) cvt(enc, enc_bf, (int)enc_elems);
  k_pad_outw<<<(VP * HH + 255) / 256, 256, 0, stream>>>(out_w, outw_bf);
  k_init_out<<<(BB * VV + 255) / 256, 256, 0, stream>>>(out_probs, out_idx);
  k_init_state<<<(BB * HH + 255) / 256, 256, 0, stream>>>(h, h_bf, idxb);

  constexpr int MH = 4;              // m-tiles per wave
  const int tM = BB / 16;            // 8
  const int mg = tM / MH;            // 2 wave groups in M
  auto gemm = [&](const bf16_t* A, int lda, const bf16_t* W, int ldw,
                  const float* bias, int biasN, float* C, int ldc, int K, int N) {
    int tN = N / 16, tot = mg * tN;
    k_gemm_xwt<MH><<<(tot * 32 + 255) / 256, 256, 0, stream>>>(
        A, lda, W, ldw, bias, biasN, C, ldc, K, tN, tot);
  };

  for (int t = 1; t < LL; ++t) {
    // th = h @ attn_w^T + attn_b     (M=128, N=1024, K=1024)
    gemm(h_bf, HH, attnw_bf, HH, attn_b, HH, th, HH, HH, HH);
    // attention + embedding -> x_bf
    if (use_enc_bf)
      k_attn_bf16<<<BB, 256, 0, stream>>>(enc_bf, th, emb_bf, idxb, x_bf);
    else
      k_attn_f32<<<BB, 256, 0, stream>>>(enc, th, emb_bf, idxb, x_bf);
    // gi = x @ w_ih^T + b_ih         (M=128, N=3072, K=2048)
    gemm(x_bf, 2 * HH, wih_bf, 2 * HH, b_ih, 3 * HH, gi, 3 * HH, 2 * HH, 3 * HH);
    // gh = h @ w_hh^T + b_hh         (M=128, N=3072, K=1024)
    gemm(h_bf, HH, whh_bf, HH, b_hh, 3 * HH, gh, 3 * HH, HH, 3 * HH);
    // GRU cell -> h, h_bf
    k_gru_cell<<<(BB * HH + 255) / 256, 256, 0, stream>>>(gi, gh, h, h_bf);
    // logits = h_new @ out_w^T + out_b   (M=128, N=5008(pad), K=1024)
    gemm(h_bf, HH, outw_bf, HH, out_b, VV, logits, VP, HH, VP);
    // softmax over V, write probs + argmax
    k_vsoftmax<<<BB, 256, 0, stream>>>(logits, out_probs, out_idx, idxb, t);
  }
}
